// MultiHeadLatentAttention_6579889897856
// MI455X (gfx1250) — matrix operand order repaired and checked by reading, not compiled, not run on hardware
//
#include <hip/hip_runtime.h>
#include <hip/hip_bf16.h>

// ---------------------------------------------------------------------------
// MLA forward for MI455X (gfx1250, wave32, WMMA).
// bf16 WMMA (fp32 accumulate) for all GEMMs; weights pre-transposed at
// conversion time so both A- and B-tile LDS fills are vectorized; epilogues
// are templated (no runtime branches in stores).
// ---------------------------------------------------------------------------

#define B_   2
#define L_   2048
#define D_   4096
#define DZ_  1024
#define H_   16
#define HD_  64
#define M_   (B_ * L_)      // 4096 rows
#define HHD  (H_ * HD_)     // 1024
#define DK_  (2 * HD_)      // 128 (per-head q/k dim)
#define SCALE_ 0.08838834764831845f   // 1/sqrt(128)

typedef __attribute__((ext_vector_type(16))) __bf16 v16bf;
typedef __attribute__((ext_vector_type(8)))  __bf16 v8bf;
typedef __attribute__((ext_vector_type(4)))  __bf16 v4bf;
typedef __attribute__((ext_vector_type(8)))  float  v8f;

__device__ __forceinline__ __bf16 f2bf(float f) { return (__bf16)f; }

__device__ __forceinline__ v16bf pack16(v8bf lo, v8bf hi) {
  v16bf r;
#pragma unroll
  for (int j = 0; j < 8; ++j) { r[j] = lo[j]; r[8 + j] = hi[j]; }
  return r;
}

__device__ __forceinline__ v8f wmma_bf16(v16bf a, v16bf b, v8f c) {
  return __builtin_amdgcn_wmma_f32_16x16x32_bf16(
      /*neg_a=*/false, a, /*neg_b=*/false, b,
      /*c_mod=*/(short)0, c, /*reuse_a=*/false, /*reuse_b=*/false);
}

// ---------------------------------------------------------------------------
// fp32 -> bf16 conversion, 4 elements/thread (row-major preserved)
// ---------------------------------------------------------------------------
__global__ void cvt_f32_to_bf16(const float* __restrict__ src,
                                __bf16* __restrict__ dst, int n4) {
  int i = blockIdx.x * blockDim.x + threadIdx.x;
  if (i >= n4) return;
  float4 v = reinterpret_cast<const float4*>(src)[i];
  v4bf o;
  o[0] = f2bf(v.x); o[1] = f2bf(v.y); o[2] = f2bf(v.z); o[3] = f2bf(v.w);
  reinterpret_cast<v4bf*>(dst)[i] = o;
}

// ---------------------------------------------------------------------------
// fp32 (KxN row-major) -> bf16 TRANSPOSED (NxK row-major), 32x32 tiles via LDS.
// Both global sides coalesced; one-time cost per weight matrix.
// ---------------------------------------------------------------------------
__global__ __launch_bounds__(256) void cvt_transpose_bf16(
    const float* __restrict__ src, __bf16* __restrict__ dst, int K, int N)
{
  __shared__ __bf16 tile[32][36];
  const int tid = threadIdx.x;
  const int k0 = blockIdx.x * 32;
  const int n0 = blockIdx.y * 32;

  {
    int row = tid >> 3;            // 0..31 (k)
    int c4  = (tid & 7) * 4;       // 0..28 (n)
    float4 v = *reinterpret_cast<const float4*>(src + (size_t)(k0 + row) * N + n0 + c4);
    v4bf o;
    o[0] = f2bf(v.x); o[1] = f2bf(v.y); o[2] = f2bf(v.z); o[3] = f2bf(v.w);
    *reinterpret_cast<v4bf*>(&tile[row][c4]) = o;
  }
  __syncthreads();
  {
    int rn = tid >> 3;             // 0..31 (n)
    int k4 = (tid & 7) * 4;        // 0..28 (k)
    v4bf o;
    o[0] = tile[k4 + 0][rn];
    o[1] = tile[k4 + 1][rn];
    o[2] = tile[k4 + 2][rn];
    o[3] = tile[k4 + 3][rn];
    *reinterpret_cast<v4bf*>(dst + (size_t)(n0 + rn) * K + k0 + k4) = o;
  }
}

// ---------------------------------------------------------------------------
// Tiled bf16 WMMA GEMM: C(MxN) = A(MxK) @ B(KxN) [+ bias]
// B is supplied PRE-TRANSPOSED: Bt is (NxK) row-major.
// Block: 256 threads (8 waves), tile 128x128, K-step 32, wave grid 2x4.
// Epilogue (compile-time): BIAS add; SF fp32 row-major; SB bf16 row-major;
//   PM==1 bf16 head-permuted [(b*H+h)*L + l]*OS + c   (Q/K)
//   PM==2 bf16 head-permuted [(b*H+h)*64 + c]*L + l   (V^T)
// ---------------------------------------------------------------------------
template <bool BIAS, bool SF, bool SB, int PM>
__global__ __launch_bounds__(256) void gemm_bf16_kernel(
    const __bf16* __restrict__ A, const __bf16* __restrict__ Bt,
    const float* __restrict__ bias,
    float* __restrict__ Cf, __bf16* __restrict__ Cb, __bf16* __restrict__ Cp,
    int M, int N, int K, int OSp)
{
  __shared__ __bf16 Als[128][40];   // A tile: [m][kk]
  __shared__ __bf16 Bls[128][40];   // B tile: [n][kk] (Bt rows are contiguous in k)

  const int tid  = threadIdx.x;
  const int wave = tid >> 5;
  const int lane = tid & 31;
  const int half = lane >> 4;
  const int l16  = lane & 15;
  const int wm   = wave >> 2;          // 0..1
  const int wn   = wave & 3;           // 0..3
  const int tm0  = blockIdx.x * 128;
  const int tn0  = blockIdx.y * 128;

  v8f acc[4][2];
#pragma unroll
  for (int mf = 0; mf < 4; ++mf)
#pragma unroll
    for (int nf = 0; nf < 2; ++nf)
#pragma unroll
      for (int r = 0; r < 8; ++r) acc[mf][nf][r] = 0.0f;

  const int trow = tid >> 1;           // 0..127
  const int tcol = (tid & 1) * 16;     // 0 or 16

  for (int k0 = 0; k0 < K; k0 += 32) {
    { // A tile: 16 contiguous bf16 per thread
      const __bf16* src = A + (size_t)(tm0 + trow) * K + (k0 + tcol);
      *(v8bf*)&Als[trow][tcol]     = *(const v8bf*)(src);
      *(v8bf*)&Als[trow][tcol + 8] = *(const v8bf*)(src + 8);
    }
    { // B tile: same vectorized copy from pre-transposed weights
      const __bf16* src = Bt + (size_t)(tn0 + trow) * K + (k0 + tcol);
      *(v8bf*)&Bls[trow][tcol]     = *(const v8bf*)(src);
      *(v8bf*)&Bls[trow][tcol + 8] = *(const v8bf*)(src + 8);
    }
    __syncthreads();

    v16bf af[4], bfr[2];
#pragma unroll
    for (int mf = 0; mf < 4; ++mf) {
      int row = wm * 64 + mf * 16 + l16;
      v8bf lo = *(const v8bf*)&Als[row][half * 8];
      v8bf hi = *(const v8bf*)&Als[row][16 + half * 8];
      af[mf] = pack16(lo, hi);
    }
#pragma unroll
    for (int nf = 0; nf < 2; ++nf) {
      int col = wn * 32 + nf * 16 + l16;
      v8bf lo = *(const v8bf*)&Bls[col][half * 8];
      v8bf hi = *(const v8bf*)&Bls[col][16 + half * 8];
      bfr[nf] = pack16(lo, hi);
    }
#pragma unroll
    for (int mf = 0; mf < 4; ++mf)
#pragma unroll
      for (int nf = 0; nf < 2; ++nf)
        acc[mf][nf] = wmma_bf16(af[mf], bfr[nf], acc[mf][nf]);

    __syncthreads();
  }

  // epilogue (fully compile-time selected)
#pragma unroll
  for (int mf = 0; mf < 4; ++mf) {
#pragma unroll
    for (int nf = 0; nf < 2; ++nf) {
#pragma unroll
      for (int r = 0; r < 8; ++r) {
        int m = tm0 + wm * 64 + mf * 16 + r + half * 8;
        int n = tn0 + wn * 32 + nf * 16 + l16;
        float v = acc[mf][nf][r];
        if (BIAS) v += bias[n];
        if (SF) Cf[(size_t)m * N + n] = v;
        if (SB) Cb[(size_t)m * N + n] = f2bf(v);
        if (PM == 1) {
          int b = m >> 11, l = m & (L_ - 1);       // L_ = 2048
          int h = n >> 6,  c = n & 63;
          Cp[((size_t)(b * H_ + h) * L_ + l) * OSp + c] = f2bf(v);
        } else if (PM == 2) {
          int b = m >> 11, l = m & (L_ - 1);
          int h = n >> 6,  c = n & 63;
          Cp[((size_t)(b * H_ + h) * 64 + c) * (size_t)L_ + l] = f2bf(v);
        }
      }
    }
  }
}

// ---------------------------------------------------------------------------
// RoPE: one thread per (b,l,h,i), i in [0,32).
// out[i]    = x[i]*cos - x[i+32]*sin
// out[i+32] = x[i+32]*cos + x[i]*sin,  ang = l * 10000^(-i/32)
// ---------------------------------------------------------------------------
__global__ void rope_kernel(const float* __restrict__ qpre,
                            const float* __restrict__ kpre,
                            __bf16* __restrict__ Qb, __bf16* __restrict__ Kb,
                            float* __restrict__ kr_out)
{
  int idx = blockIdx.x * blockDim.x + threadIdx.x;
  if (idx >= B_ * L_ * H_ * 32) return;
  int i = idx & 31;
  int h = (idx >> 5) & (H_ - 1);
  int l = (idx >> 9) & (L_ - 1);
  int b = idx >> 20;

  float inv = __expf(-(float)i * (9.210340371976184f / 32.0f));
  float ang = (float)l * inv;
  float s, c;
  __sincosf(ang, &s, &c);

  size_t base = ((size_t)b * L_ + l) * HHD + h * HD_;
  float q1 = qpre[base + i], q2 = qpre[base + 32 + i];
  float k1 = kpre[base + i], k2 = kpre[base + 32 + i];
  float oq1 = q1 * c - q2 * s, oq2 = q2 * c + q1 * s;
  float ok1 = k1 * c - k2 * s, ok2 = k2 * c + k1 * s;

  size_t qk = ((size_t)(b * H_ + h) * L_ + l) * DK_ + 64;
  Qb[qk + i] = f2bf(oq1); Qb[qk + 32 + i] = f2bf(oq2);
  Kb[qk + i] = f2bf(ok1); Kb[qk + 32 + i] = f2bf(ok2);
  kr_out[base + i] = ok1; kr_out[base + 32 + i] = ok2;
}

// ---------------------------------------------------------------------------
// Fused attention per (b,h, 128-q-row tile).
// S = Q@K^T (WMMA over d=128), scale, causal mask, write scores (fp32),
// online row max/sumexp, head_out += S @ V (reference uses raw masked
// scores, not softmax) via LDS round-trip to re-layout S as A-fragments.
// ---------------------------------------------------------------------------
__global__ __launch_bounds__(256) void attn_kernel(
    const __bf16* __restrict__ Qb, const __bf16* __restrict__ Kb,
    const __bf16* __restrict__ Vt,
    float* __restrict__ scores_out, float* __restrict__ rowmax,
    float* __restrict__ rowsum, float* __restrict__ ho)
{
  const int bh = blockIdx.x;           // b*H + h
  const int q0 = blockIdx.y * 128;
  const int b  = bh >> 4;
  const int h  = bh & 15;

  const int tid  = threadIdx.x;
  const int wave = tid >> 5;
  const int lane = tid & 31;
  const int half = lane >> 4;
  const int l16  = lane & 15;

  const __bf16* Qrow = Qb + (size_t)bh * L_ * DK_;
  const __bf16* Krow = Kb + (size_t)bh * L_ * DK_;
  const __bf16* Vrow = Vt + (size_t)bh * 64 * L_;
  float* Sout = scores_out + (size_t)bh * L_ * L_;

  // cache Q A-fragments for this wave's 16 rows (4 d-steps of 32)
  v16bf qf[4];
  {
    int qm = q0 + wave * 16 + l16;
#pragma unroll
    for (int di = 0; di < 4; ++di) {
      const __bf16* p = Qrow + (size_t)qm * DK_ + di * 32;
      v8bf lo = *(const v8bf*)(p + half * 8);
      v8bf hi = *(const v8bf*)(p + 16 + half * 8);
      qf[di] = pack16(lo, hi);
    }
  }

  __shared__ __bf16 Sls[8][16][136];   // per-wave 16x128 score tile (bf16)

  v8f hacc[4];
#pragma unroll
  for (int nf = 0; nf < 4; ++nf)
#pragma unroll
    for (int r = 0; r < 8; ++r) hacc[nf][r] = 0.0f;

  float rmax[8], rsum[8];
#pragma unroll
  for (int r = 0; r < 8; ++r) { rmax[r] = -3.0e38f; rsum[r] = 0.0f; }

  for (int kt = 0; kt < L_ / 128; ++kt) {
    const int k0 = kt * 128;

    v8f sacc[8];
#pragma unroll
    for (int t = 0; t < 8; ++t)
#pragma unroll
      for (int r = 0; r < 8; ++r) sacc[t][r] = 0.0f;

#pragma unroll
    for (int di = 0; di < 4; ++di) {
#pragma unroll
      for (int t = 0; t < 8; ++t) {
        int key = k0 + t * 16 + l16;
        const __bf16* p = Krow + (size_t)key * DK_ + di * 32 + half * 8;
        v16bf kb = pack16(*(const v8bf*)p, *(const v8bf*)(p + 16));
        sacc[t] = wmma_bf16(qf[di], kb, sacc[t]);
      }
    }

    // scale, mask, write scores, stash bf16 tile in LDS, per-row tile max
    float tmax[8];
#pragma unroll
    for (int r = 0; r < 8; ++r) tmax[r] = -3.0e38f;
#pragma unroll
    for (int t = 0; t < 8; ++t) {
      int key = k0 + t * 16 + l16;
#pragma unroll
      for (int r = 0; r < 8; ++r) {
        int q = q0 + wave * 16 + r + half * 8;
        float v = sacc[t][r] * SCALE_;
        if (key > q) v = -1000000.0f;
        sacc[t][r] = v;
        Sout[(size_t)q * L_ + key] = v;
        Sls[wave][r + half * 8][t * 16 + l16] = f2bf(v);
        tmax[r] = fmaxf(tmax[r], v);
      }
    }
    // reduce across the 16 lanes of this half-wave (cols), update stats
#pragma unroll
    for (int r = 0; r < 8; ++r) {
#pragma unroll
      for (int m = 1; m < 16; m <<= 1)
        tmax[r] = fmaxf(tmax[r], __shfl_xor(tmax[r], m, 16));
      float nm = fmaxf(rmax[r], tmax[r]);
      float part = 0.0f;
#pragma unroll
      for (int t = 0; t < 8; ++t) part += __expf(sacc[t][r] - nm);
#pragma unroll
      for (int m = 1; m < 16; m <<= 1)
        part += __shfl_xor(part, m, 16);
      rsum[r] = rsum[r] * __expf(rmax[r] - nm) + part;
      rmax[r] = nm;
    }

    // head_out += S(16x128) @ V(128x64); A-frags from own LDS region
#pragma unroll
    for (int di = 0; di < 4; ++di) {
      const __bf16* sp = &Sls[wave][l16][di * 32];
      v16bf sa = pack16(*(const v8bf*)(sp + half * 8),
                        *(const v8bf*)(sp + 16 + half * 8));
#pragma unroll
      for (int nf = 0; nf < 4; ++nf) {
        int c = nf * 16 + l16;
        const __bf16* vp = Vrow + (size_t)c * L_ + k0 + di * 32 + half * 8;
        v16bf vb = pack16(*(const v8bf*)vp, *(const v8bf*)(vp + 16));
        hacc[nf] = wmma_bf16(sa, vb, hacc[nf]);
      }
    }
  }

  // write head_out in (B, L, H*HD) layout
#pragma unroll
  for (int nf = 0; nf < 4; ++nf) {
#pragma unroll
    for (int r = 0; r < 8; ++r) {
      int q = q0 + wave * 16 + r + half * 8;
      int c = nf * 16 + l16;
      ho[((size_t)b * L_ + q) * HHD + h * HD_ + c] = hacc[nf][r];
    }
  }
  // row stats (each 16-lane group holds the full reduction)
  if (l16 == 0) {
#pragma unroll
    for (int r = 0; r < 8; ++r) {
      int q = q0 + wave * 16 + r + half * 8;
      rowmax[(size_t)bh * L_ + q] = rmax[r];
      rowsum[(size_t)bh * L_ + q] = rsum[r];
    }
  }
}

// ---------------------------------------------------------------------------
// attn_weights = exp(scores - rowmax) / rowsum   (4 elements/thread)
// ---------------------------------------------------------------------------
__global__ void softmax_fin_kernel(const float* __restrict__ scores,
                                   const float* __restrict__ rowmax,
                                   const float* __restrict__ rowsum,
                                   float* __restrict__ attn, int n4)
{
  int i = blockIdx.x * blockDim.x + threadIdx.x;
  if (i >= n4) return;
  size_t base = (size_t)i * 4;
  size_t row = base / L_;            // = bh*L + q
  float m = rowmax[row];
  float inv = 1.0f / rowsum[row];
  float4 s = reinterpret_cast<const float4*>(scores)[i];
  float4 o;
  o.x = __expf(s.x - m) * inv;
  o.y = __expf(s.y - m) * inv;
  o.z = __expf(s.z - m) * inv;
  o.w = __expf(s.w - m) * inv;
  reinterpret_cast<float4*>(attn)[i] = o;
}

// ---------------------------------------------------------------------------
// Host-side orchestration
// ---------------------------------------------------------------------------
extern "C" void kernel_launch(void* const* d_in, const int* in_sizes, int n_in,
                              void* d_out, int out_size, void* d_ws, size_t ws_size,
                              hipStream_t stream) {
  (void)in_sizes; (void)n_in; (void)out_size; (void)ws_size;

  const float* x        = (const float*)d_in[0];
  const float* W_latent = (const float*)d_in[1];
  const float* W_q_down = (const float*)d_in[2];
  const float* b_q_down = (const float*)d_in[3];
  const float* W_q_up   = (const float*)d_in[4];
  const float* W_k_up   = (const float*)d_in[5];
  const float* W_v_up   = (const float*)d_in[6];
  const float* W_x_rope = (const float*)d_in[7];
  const float* W_k_rope = (const float*)d_in[8];
  const float* W_o      = (const float*)d_in[9];

  float* out    = (float*)d_out;                       // (B,L,D)
  float* z_out  = out    + (size_t)M_ * D_;            // (B,L,DZ)
  float* kr_out = z_out  + (size_t)M_ * DZ_;           // (B,L,H*HD)
  float* aw_out = kr_out + (size_t)M_ * HHD;           // (B,H,L,L)
  float* sc_out = aw_out + (size_t)B_ * H_ * L_ * L_;  // (B,H,L,L)

  // workspace bump allocator (256B aligned)
  char* wsb = (char*)d_ws;
  size_t off = 0;
  auto alloc = [&](size_t bytes) -> void* {
    void* p = wsb + off;
    off += (bytes + 255) & ~(size_t)255;
    return p;
  };

  __bf16* xb     = (__bf16*)alloc((size_t)M_ * D_ * 2);
  __bf16* Wlat_t = (__bf16*)alloc((size_t)D_ * DZ_ * 2);   // transposed (N x K)
  __bf16* Wqd_t  = (__bf16*)alloc((size_t)D_ * DZ_ * 2);
  __bf16* Wkr_t  = (__bf16*)alloc((size_t)D_ * HHD * 2);
  __bf16* Wqu_t  = (__bf16*)alloc((size_t)DZ_ * HHD * 2);
  __bf16* Wku_t  = (__bf16*)alloc((size_t)DZ_ * HHD * 2);
  __bf16* Wvu_t  = (__bf16*)alloc((size_t)DZ_ * HHD * 2);
  __bf16* Wxr_t  = (__bf16*)alloc((size_t)DZ_ * HHD * 2);
  __bf16* Wo_t   = (__bf16*)alloc((size_t)HHD * D_ * 2);
  __bf16* zb     = (__bf16*)alloc((size_t)M_ * DZ_ * 2);
  float*  qlat_f = (float*)alloc((size_t)M_ * DZ_ * 4);
  __bf16* qlat_b = (__bf16*)alloc((size_t)M_ * DZ_ * 2);
  float*  qrp_f  = (float*)alloc((size_t)M_ * HHD * 4);
  float*  krp_f  = (float*)alloc((size_t)M_ * HHD * 4);
  __bf16* Qb     = (__bf16*)alloc((size_t)B_ * H_ * L_ * DK_ * 2);
  __bf16* Kb     = (__bf16*)alloc((size_t)B_ * H_ * L_ * DK_ * 2);
  __bf16* Vt     = (__bf16*)alloc((size_t)B_ * H_ * 64 * L_ * 2);
  float*  rmax   = (float*)alloc((size_t)B_ * H_ * L_ * 4);
  float*  rsum   = (float*)alloc((size_t)B_ * H_ * L_ * 4);
  float*  ho_f   = (float*)alloc((size_t)M_ * HHD * 4);
  __bf16* ho_b   = (__bf16*)alloc((size_t)M_ * HHD * 2);

  auto cvt = [&](const float* s, __bf16* d, size_t n) {
    int n4 = (int)(n / 4);
    cvt_f32_to_bf16<<<dim3((n4 + 255) / 256), dim3(256), 0, stream>>>(s, d, n4);
  };
  auto cvtT = [&](const float* s, __bf16* d, int K, int N) {
    cvt_transpose_bf16<<<dim3(K / 32, N / 32), dim3(256), 0, stream>>>(s, d, K, N);
  };

  // 1) convert inputs; weights converted + transposed (NxK)
  cvt(x, xb, (size_t)M_ * D_);
  cvtT(W_latent, Wlat_t, D_, DZ_);
  cvtT(W_q_down, Wqd_t,  D_, DZ_);
  cvtT(W_k_rope, Wkr_t,  D_, HHD);
  cvtT(W_q_up,   Wqu_t,  DZ_, HHD);
  cvtT(W_k_up,   Wku_t,  DZ_, HHD);
  cvtT(W_v_up,   Wvu_t,  DZ_, HHD);
  cvtT(W_x_rope, Wxr_t,  DZ_, HHD);
  cvtT(W_o,      Wo_t,   HHD, D_);

  // 2) stage-1 projections from x (K = 4096)
  gemm_bf16_kernel<false, true, true, 0><<<dim3(M_/128, DZ_/128), dim3(256), 0, stream>>>(
      xb, Wlat_t, nullptr, z_out, zb, nullptr, M_, DZ_, D_, 0);            // z (+bf16)
  gemm_bf16_kernel<true, true, true, 0><<<dim3(M_/128, DZ_/128), dim3(256), 0, stream>>>(
      xb, Wqd_t, b_q_down, qlat_f, qlat_b, nullptr, M_, DZ_, D_, 0);       // q_lat (+bias,+bf16)
  gemm_bf16_kernel<false, true, false, 0><<<dim3(M_/128, HHD/128), dim3(256), 0, stream>>>(
      xb, Wkr_t, nullptr, krp_f, nullptr, nullptr, M_, HHD, D_, 0);        // k_rope pre

  // 3) stage-2 projections (K = 1024), stored in per-head layouts
  gemm_bf16_kernel<false, false, false, 1><<<dim3(M_/128, HHD/128), dim3(256), 0, stream>>>(
      qlat_b, Wqu_t, nullptr, nullptr, nullptr, Qb, M_, HHD, DZ_, DK_);    // query -> Q[:, :64]
  gemm_bf16_kernel<false, true, false, 0><<<dim3(M_/128, HHD/128), dim3(256), 0, stream>>>(
      qlat_b, Wxr_t, nullptr, qrp_f, nullptr, nullptr, M_, HHD, DZ_, 0);   // q_rope pre
  gemm_bf16_kernel<false, false, false, 1><<<dim3(M_/128, HHD/128), dim3(256), 0, stream>>>(
      zb, Wku_t, nullptr, nullptr, nullptr, Kb, M_, HHD, DZ_, DK_);        // key_nope -> K[:, :64]
  gemm_bf16_kernel<false, false, false, 2><<<dim3(M_/128, HHD/128), dim3(256), 0, stream>>>(
      zb, Wvu_t, nullptr, nullptr, nullptr, Vt, M_, HHD, DZ_, 0);          // value -> V^T

  // 4) RoPE: fill Q/K rope halves (cols 64..127) + fp32 key_rope output
  {
    int n = B_ * L_ * H_ * 32;
    rope_kernel<<<dim3((n + 255) / 256), dim3(256), 0, stream>>>(
        qrp_f, krp_f, Qb, Kb, kr_out);
  }

  // 5) fused attention: scores + online softmax stats + head_out
  attn_kernel<<<dim3(B_ * H_, L_ / 128), dim3(256), 0, stream>>>(
      Qb, Kb, Vt, sc_out, rmax, rsum, ho_f);

  // 6) output projection: out = head_out @ W_o  (K = 1024, N = 4096)
  cvt(ho_f, ho_b, (size_t)M_ * HHD);
  gemm_bf16_kernel<false, true, false, 0><<<dim3(M_/128, D_/128), dim3(256), 0, stream>>>(
      ho_b, Wo_t, nullptr, out, nullptr, nullptr, M_, D_, HHD, 0);

  // 7) finalize attn_weights = softmax(scores)
  {
    size_t total = (size_t)B_ * H_ * L_ * L_;
    int n4 = (int)(total / 4);
    softmax_fin_kernel<<<dim3((n4 + 255) / 256), dim3(256), 0, stream>>>(
        sc_out, rmax, rsum, aw_out, n4);
  }
}
